// HGT_88527865905251
// MI455X (gfx1250) — compile-verified, matrix operand order checked by reading, not run on hardware
//
#include <hip/hip_runtime.h>

// ---------------------------------------------------------------------------
// HGT forward for MI455X (gfx1250, wave32, WMMA).
// Dense parts: bf16 WMMA (v_wmma_f32_16x16x32_bf16) with f32 accumulate.
//   Weights are pre-swizzled into WMMA B-fragment order once (tiny kernels),
//   so the GEMM stages them to LDS with a straight async b128 block copy
//   (global_load_async_to_lds_b128 + s_wait_asynccnt when available).
// Sparse parts: gather + segment-softmax via ordered-uint atomicMax and
// global_atomic_add_f32 scatter.
// ---------------------------------------------------------------------------

typedef __attribute__((ext_vector_type(16))) __bf16        v16bf;
typedef __attribute__((ext_vector_type(8)))  float         v8f;
typedef __attribute__((ext_vector_type(4)))  unsigned int  v4u;
typedef __attribute__((ext_vector_type(4)))  int           v4i;

union Frag16 { v4u u[2]; v16bf v; };   // 16 bf16 = 32 bytes

#if defined(__HIP_DEVICE_COMPILE__) && __has_builtin(__builtin_amdgcn_global_load_async_to_lds_b128)
#define HAVE_ASYNC_LDS 1
#endif
#define AS1 __attribute__((address_space(1)))
#define AS3 __attribute__((address_space(3)))

__device__ __forceinline__ unsigned short f2bf(float f) {
    unsigned int u = __float_as_uint(f);
    u += 0x7FFFu + ((u >> 16) & 1u);          // round-to-nearest-even
    return (unsigned short)(u >> 16);
}
// monotone float <-> uint map so unsigned atomicMax orders like float max
__device__ __forceinline__ unsigned int ford(float f) {
    unsigned int u = __float_as_uint(f);
    return (u & 0x80000000u) ? ~u : (u | 0x80000000u);
}
__device__ __forceinline__ float dord(unsigned int u) {
    return __uint_as_float((u & 0x80000000u) ? (u & 0x7FFFFFFFu) : ~u);
}

// fragment-order index mapping for a 64x64 bf16 matrix packed as 2048 dwords:
// dword d -> fragment f=d>>3 (t=f>>6 n-tile, s=(f>>5)&1 k-step, L=f&31 lane),
// pair p=d&7 holding bf16 elements j0=2p, j1=2p+1 of the lane's 16 values.
__device__ __forceinline__ void frag_kc(int d, int& k0, int& k1, int& c) {
    int f = d >> 3, p = d & 7;
    int t = f >> 6, s = (f >> 5) & 1, L = f & 31;
    c  = t * 16 + (L & 15);
    int kb = s * 32 + ((L >> 4) << 3);
    int j0 = 2 * p, j1 = j0 + 1;
    k0 = kb + (j0 < 8 ? j0 : j0 + 8);   // j>=8 -> kb + 16 + (j-8)
    k1 = kb + (j1 < 8 ? j1 : j1 + 8);
}

// ---------------------------------------------------------------------------
// GEMM: C[N x 64] = A_bf16[N x 64] * Wswz (+bias) (+relu)
// Wswz: 64x64 bf16 weight already in B-fragment order (2048 dwords).
// Block = 256 threads = 8 waves; wave computes 16 rows x 64 cols
// = 4 n-tiles x 2 k-steps of v_wmma_f32_16x16x32_bf16.
// ---------------------------------------------------------------------------
__global__ __launch_bounds__(256)
void gemm_bf16_n64(const unsigned short* __restrict__ A, int N,
                   const unsigned int* __restrict__ Wswz,
                   const float* __restrict__ bias,
                   float* __restrict__ Cf, unsigned short* __restrict__ Cb,
                   int relu)
{
    __shared__ unsigned int sW[2048];   // 8KB staged weight

    {   // straight block copy: 256 threads x 2 x b128
        int tid = threadIdx.x;
#if defined(HAVE_ASYNC_LDS)
        __builtin_amdgcn_global_load_async_to_lds_b128(
            (AS1 v4i*)(Wswz + tid * 4),
            (AS3 v4i*)(sW + tid * 4), 0, 0);
        __builtin_amdgcn_global_load_async_to_lds_b128(
            (AS1 v4i*)(Wswz + 1024 + tid * 4),
            (AS3 v4i*)(sW + 1024 + tid * 4), 0, 0);
#if __has_builtin(__builtin_amdgcn_s_wait_asynccnt)
        __builtin_amdgcn_s_wait_asynccnt(0);
#else
        asm volatile("s_wait_asynccnt 0x0" ::: "memory");
#endif
#else
        ((v4u*)sW)[tid]       = ((const v4u*)Wswz)[tid];
        ((v4u*)sW)[tid + 256] = ((const v4u*)Wswz)[tid + 256];
#endif
    }
    __syncthreads();

    int wave = threadIdx.x >> 5;
    int lane = threadIdx.x & 31;
    int rowBase = blockIdx.x * 128 + wave * 16;
    if (rowBase >= N) return;

    // A fragments for both k-steps (16x32 bf16 each)
    int r = rowBase + (lane & 15);
    if (r >= N) r = N - 1;                       // clamp (stores are guarded)
    const unsigned short* Ap = A + (size_t)r * 64;
    int klane = (lane >> 4) << 3;                // 0 or 8
    Frag16 a0, a1;
    a0.u[0] = *(const v4u*)(Ap + klane);
    a0.u[1] = *(const v4u*)(Ap + klane + 16);
    a1.u[0] = *(const v4u*)(Ap + 32 + klane);
    a1.u[1] = *(const v4u*)(Ap + 32 + klane + 16);

    v8f acc[4];
    v8f z = {0.f, 0.f, 0.f, 0.f, 0.f, 0.f, 0.f, 0.f};
#pragma unroll
    for (int t = 0; t < 4; ++t) acc[t] = z;

#pragma unroll
    for (int t = 0; t < 4; ++t) {
        Frag16 b0, b1;
        const v4u* sp0 = (const v4u*)&sW[(((t * 2 + 0) * 32) + lane) * 8];
        b0.u[0] = sp0[0]; b0.u[1] = sp0[1];
        acc[t] = __builtin_amdgcn_wmma_f32_16x16x32_bf16(
            false, a0.v, false, b0.v, (short)0, acc[t], false, false);
        const v4u* sp1 = (const v4u*)&sW[(((t * 2 + 1) * 32) + lane) * 8];
        b1.u[0] = sp1[0]; b1.u[1] = sp1[1];
        acc[t] = __builtin_amdgcn_wmma_f32_16x16x32_bf16(
            false, a1.v, false, b1.v, (short)0, acc[t], false, false);
    }

    // epilogue: C layout = lane (L&15) -> col, vgpr i + (L>>4)*8 -> row
    int colL = lane & 15;
    int rOff = (lane >> 4) << 3;
    if (rowBase + 16 <= N) {                     // full-strip fast path
#pragma unroll
        for (int t = 0; t < 4; ++t) {
            int c = t * 16 + colL;
            float bv = bias ? bias[c] : 0.f;
#pragma unroll
            for (int i = 0; i < 8; ++i) {
                size_t rr = (size_t)(rowBase + rOff + i);
                float v = acc[t][i] + bv;
                if (relu) v = fmaxf(v, 0.f);
                if (Cf) Cf[rr * 64 + c] = v;
                if (Cb) Cb[rr * 64 + c] = f2bf(v);
            }
        }
    } else {
#pragma unroll
        for (int t = 0; t < 4; ++t) {
            int c = t * 16 + colL;
            float bv = bias ? bias[c] : 0.f;
#pragma unroll
            for (int i = 0; i < 8; ++i) {
                int rr = rowBase + rOff + i;
                if (rr < N) {
                    float v = acc[t][i] + bv;
                    if (relu) v = fmaxf(v, 0.f);
                    if (Cf) Cf[(size_t)rr * 64 + c] = v;
                    if (Cb) Cb[(size_t)rr * 64 + c] = f2bf(v);
                }
            }
        }
    }
}

// ---------------------------------------------------------------------------
// small utility kernels
// ---------------------------------------------------------------------------
__global__ void k_zero(unsigned int* __restrict__ p, int n) {
    int i = blockIdx.x * blockDim.x + threadIdx.x;
    if (i < n) p[i] = 0u;
}

// f32 64x64 weight -> bf16 in B-fragment order (2048 dwords)
__global__ void k_cvt_swz(const float* __restrict__ W, unsigned int* __restrict__ o) {
    int d = blockIdx.x * blockDim.x + threadIdx.x;
    if (d >= 2048) return;
    int k0, k1, c;
    frag_kc(d, k0, k1, c);
    unsigned int lo = f2bf(W[k0 * 64 + c]);
    unsigned int hi = f2bf(W[k1 * 64 + c]);
    o[d] = lo | (hi << 16);
}

// rel[2,32,32] -> block-diagonal 64x64 bf16 in B-fragment order
// BD[h*32+d][h*32+e] = rel[h,d,e]
__global__ void k_build_bd_swz(const float* __restrict__ rel, unsigned int* __restrict__ o) {
    int d = blockIdx.x * blockDim.x + threadIdx.x;
    if (d >= 2048) return;
    int k0, k1, c;
    frag_kc(d, k0, k1, c);
    int ch = c >> 5;
    float f0 = ((k0 >> 5) == ch) ? rel[ch * 1024 + (k0 & 31) * 32 + (c & 31)] : 0.f;
    float f1 = ((k1 >> 5) == ch) ? rel[ch * 1024 + (k1 & 31) * 32 + (c & 31)] : 0.f;
    o[d] = (unsigned int)f2bf(f0) | ((unsigned int)f2bf(f1) << 16);
}

// input projection: h = relu(x @ W + b), K in {10,8,6}
__global__ void k_in_proj(const float* __restrict__ x, const float* __restrict__ W,
                          const float* __restrict__ b, int N, int K,
                          float* __restrict__ hf, unsigned short* __restrict__ hb)
{
    int i = blockIdx.x * blockDim.x + threadIdx.x;
    if (i >= N * 64) return;
    int n = i >> 6, c = i & 63;
    float acc = b[c];
    const float* xr = x + (size_t)n * K;
    for (int k = 0; k < K; ++k) acc += xr[k] * W[k * 64 + c];
    acc = fmaxf(acc, 0.f);
    hf[i] = acc;
    hb[i] = f2bf(acc);
}

// phase 1: raw scores + segment max (ordered-uint atomicMax)
__global__ void k_edge_score(const int* __restrict__ ei, int E,
                             const float* __restrict__ q, const float* __restrict__ ke,
                             const float* __restrict__ prel,
                             unsigned int* __restrict__ segmax, float* __restrict__ sc)
{
    int idx = blockIdx.x * blockDim.x + threadIdx.x;
    if (idx >= E * 2) return;
    int e = idx >> 1, h = idx & 1;
    int si = ei[e], di = ei[E + e];
    const float* qp = q  + (size_t)di * 64 + h * 32;
    const float* kp = ke + (size_t)si * 64 + h * 32;
    float acc = 0.f;
#pragma unroll
    for (int d = 0; d < 32; ++d) acc += qp[d] * kp[d];
    acc *= prel[h] * 0.17677669529663687f;   // 1/sqrt(DH=32)
    sc[idx] = acc;
    atomicMax(&segmax[di * 2 + h], ford(acc));
}

// phase 2: e = exp(sc - max); accumulate denominator
__global__ void k_edge_exp(const int* __restrict__ ei, int E,
                           const unsigned int* __restrict__ segmax,
                           float* __restrict__ den, float* __restrict__ sc)
{
    int idx = blockIdx.x * blockDim.x + threadIdx.x;
    if (idx >= E * 2) return;
    int e = idx >> 1, h = idx & 1;
    int di = ei[E + e];
    float v = expf(sc[idx] - dord(segmax[di * 2 + h]));
    sc[idx] = v;
    atomicAdd(&den[di * 2 + h], v);
}

// phase 3: alpha-weighted scatter of ve into agg
__global__ void k_edge_agg(const int* __restrict__ ei, int E,
                           const float* __restrict__ den, const float* __restrict__ sc,
                           const float* __restrict__ ve, float* __restrict__ agg)
{
    int idx = blockIdx.x * blockDim.x + threadIdx.x;
    if (idx >= E * 2) return;
    int e = idx >> 1, h = idx & 1;
    int si = ei[e], di = ei[E + e];
    float alpha = sc[idx] / (den[di * 2 + h] + 1e-16f);
    const float* vp = ve + (size_t)si * 64 + h * 32;
    float* ap = agg + (size_t)di * 64 + h * 32;
#pragma unroll
    for (int d = 0; d < 32; ++d) atomicAdd(ap + d, alpha * vp[d]);
}

// exact gelu -> bf16
__global__ void k_gelu(const float* __restrict__ x, unsigned short* __restrict__ o, int n) {
    int i = blockIdx.x * blockDim.x + threadIdx.x;
    if (i >= n) return;
    float v = x[i];
    o[i] = f2bf(0.5f * v * (1.f + erff(v * 0.70710678118654752f)));
}

// h = relu(LN((2-g)*h + g*o)); one wave32 per node, 2 cols per lane
__global__ void k_combine_ln(float* __restrict__ h, unsigned short* __restrict__ hb,
                             const float* __restrict__ o, const float* __restrict__ skip,
                             const float* __restrict__ lng, const float* __restrict__ lnb, int N)
{
    int gid = blockIdx.x * blockDim.x + threadIdx.x;
    int node = gid >> 5;
    int lane = threadIdx.x & 31;
    if (node >= N) return;
    float g = 1.f / (1.f + expf(-skip[0]));
    size_t base = (size_t)node * 64;
    float v0 = (2.f - g) * h[base + lane]      + g * o[base + lane];
    float v1 = (2.f - g) * h[base + lane + 32] + g * o[base + lane + 32];
    float s  = v0 + v1;
    float s2 = v0 * v0 + v1 * v1;
#pragma unroll
    for (int m = 16; m >= 1; m >>= 1) {
        s  += __shfl_xor(s,  m, 32);
        s2 += __shfl_xor(s2, m, 32);
    }
    float mean = s * (1.f / 64.f);
    float var  = s2 * (1.f / 64.f) - mean * mean;
    float inv  = rsqrtf(var + 1e-5f);
    float y0 = fmaxf(lng[lane]      * (v0 - mean) * inv + lnb[lane],      0.f);
    float y1 = fmaxf(lng[lane + 32] * (v1 - mean) * inv + lnb[lane + 32], 0.f);
    h[base + lane]       = y0;  h[base + lane + 32]  = y1;
    hb[base + lane]      = f2bf(y0);
    hb[base + lane + 32] = f2bf(y1);
}

// final 64 -> 1 projection
__global__ void k_head2(const float* __restrict__ hh, const float* __restrict__ w2,
                        const float* __restrict__ b2, float* __restrict__ out, int N)
{
    int n = blockIdx.x * blockDim.x + threadIdx.x;
    if (n >= N) return;
    const float* hp = hh + (size_t)n * 64;
    float acc = b2[0];
#pragma unroll
    for (int k = 0; k < 64; ++k) acc += hp[k] * w2[k];
    out[n] = acc;
}

// ---------------------------------------------------------------------------
static inline int divUp(int a, int b) { return (a + b - 1) / b; }

extern "C" void kernel_launch(void* const* d_in, const int* in_sizes, int n_in,
                              void* d_out, int out_size, void* d_ws, size_t ws_size,
                              hipStream_t stream)
{
    (void)in_sizes; (void)n_in; (void)out_size; (void)ws_size;

    const int NF = 200000, NA = 3000, NL = 300;
    const int Nn[3]  = {NF, NA, NL};          // flight, airport, airline
    const int Kin[3] = {10, 8, 6};

    const float* xin[3] = {(const float*)d_in[0], (const float*)d_in[1], (const float*)d_in[2]};
    const int*   ei[4]  = {(const int*)d_in[3], (const int*)d_in[4],
                           (const int*)d_in[5], (const int*)d_in[6]};
    const int Ecnt[4]   = {400000, 400000, 200000, 200000}; // dep, rev_dep, by, rev_by

    auto L = [&](int i) { return (const float*)d_in[i]; };

    // --- param pytree leaf indices (JAX: dict keys sorted, tuples in order) ---
    const int head1W = 7, head1b = 8, head2W = 9, head2b = 10;
    const int inW[3] = {15, 13, 11}, inB[3] = {16, 14, 12};      // flight, airport, airline
    const int lnG[3] = {99, 97, 95}, lnBt[3] = {100, 98, 96};
    const int layerBase[2] = {17, 56};                            // 39 leaves each
    const int aW[3] = {4, 2, 0},    aB[3] = {5, 3, 1};
    const int kW[3] = {14, 12, 10}, kB[3] = {15, 13, 11};
    const int qW[3] = {28, 26, 24}, qB[3] = {29, 27, 25};
    const int vW[3] = {37, 35, 33}, vB[3] = {38, 36, 34};
    const int skipO[3] = {32, 31, 30};
    const int arelO[4] = {7, 9, 6, 8};      // dep, rev_dep, by, rev_by
    const int mrelO[4] = {17, 19, 16, 18};
    const int prelO[4] = {21, 23, 20, 22};

    // --- workspace carve (all accumulators re-zeroed below: capture-safe) ---
    char* cur = (char*)d_ws;
    auto alloc = [&](size_t bytes) -> char* {
        char* p = cur; cur += (bytes + 255) & ~(size_t)255; return p;
    };
    float* hf[3]; unsigned short* hb[3]; float* qd[3];
    unsigned short* kb[3]; unsigned short* vb[3];
    for (int t = 0; t < 3; ++t) hf[t] = (float*)alloc((size_t)Nn[t] * 64 * 4);
    for (int t = 0; t < 3; ++t) hb[t] = (unsigned short*)alloc((size_t)Nn[t] * 64 * 2);
    for (int t = 0; t < 3; ++t) qd[t] = (float*)alloc((size_t)Nn[t] * 64 * 4);
    for (int t = 0; t < 3; ++t) kb[t] = (unsigned short*)alloc((size_t)Nn[t] * 64 * 2);
    for (int t = 0; t < 3; ++t) vb[t] = (unsigned short*)alloc((size_t)Nn[t] * 64 * 2);
    float* big0  = (float*)alloc((size_t)NF * 64 * 4);   // ke scratch, later 'o' buffer
    float* veS   = (float*)alloc((size_t)NF * 64 * 4);   // ve regions (packed per dst)
    float* agg   = (float*)alloc((size_t)NF * 64 * 4);
    unsigned short* geluB = (unsigned short*)alloc((size_t)NF * 64 * 2);
    float* scB   = (float*)alloc((size_t)1200000 * 4);   // max dst edge group * heads
    unsigned int* segmax = (unsigned int*)alloc((size_t)NF * 2 * 4);
    float* den   = (float*)alloc((size_t)NF * 2 * 4);
    unsigned int* wslot = (unsigned int*)alloc((size_t)20 * 2048 * 4);
    auto WS = [&](int s) { return wslot + (size_t)s * 2048; };
    // slots: 0..2 Wk, 3..5 Wq, 6..8 Wv, 9..11 Wa, 12..15 BD(arel), 16..19 BD(mrel)

    dim3 B(256);

    // ---- input projections ----
    for (int t = 0; t < 3; ++t)
        k_in_proj<<<divUp(Nn[t] * 64, 256), B, 0, stream>>>(
            xin[t], L(inW[t]), L(inB[t]), Nn[t], Kin[t], hf[t], hb[t]);

    const int dstEdges[3][2] = {{1, 3}, {0, -1}, {2, -1}};   // incoming edge types per dst
    const int srcOf[4] = {0, 1, 0, 2};                       // src node type per edge type

    for (int ly = 0; ly < 2; ++ly) {
        int base = layerBase[ly];

        // weight conversion (pre-swizzled) + block-diagonal relation matrices
        for (int t = 0; t < 3; ++t) {
            k_cvt_swz<<<8, B, 0, stream>>>(L(base + kW[t]), WS(0 + t));
            k_cvt_swz<<<8, B, 0, stream>>>(L(base + qW[t]), WS(3 + t));
            k_cvt_swz<<<8, B, 0, stream>>>(L(base + vW[t]), WS(6 + t));
            k_cvt_swz<<<8, B, 0, stream>>>(L(base + aW[t]), WS(9 + t));
        }
        for (int r = 0; r < 4; ++r) {
            k_build_bd_swz<<<8, B, 0, stream>>>(L(base + arelO[r]), WS(12 + r));
            k_build_bd_swz<<<8, B, 0, stream>>>(L(base + mrelO[r]), WS(16 + r));
        }

        // K/Q/V projections (WMMA GEMMs)
        for (int t = 0; t < 3; ++t) {
            gemm_bf16_n64<<<divUp(Nn[t], 128), B, 0, stream>>>(
                hb[t], Nn[t], WS(0 + t), L(base + kB[t]), nullptr, kb[t], 0);
            gemm_bf16_n64<<<divUp(Nn[t], 128), B, 0, stream>>>(
                hb[t], Nn[t], WS(3 + t), L(base + qB[t]), qd[t], nullptr, 0);
            gemm_bf16_n64<<<divUp(Nn[t], 128), B, 0, stream>>>(
                hb[t], Nn[t], WS(6 + t), L(base + vB[t]), nullptr, vb[t], 0);
        }

        // per-destination-type attention
        for (int d = 0; d < 3; ++d) {
            int Nd = Nn[d];
            k_zero<<<divUp(Nd * 2, 256), B, 0, stream>>>(segmax, Nd * 2);
            k_zero<<<divUp(Nd * 2, 256), B, 0, stream>>>((unsigned int*)den, Nd * 2);
            k_zero<<<divUp(Nd * 64, 256), B, 0, stream>>>((unsigned int*)agg, Nd * 64);

            int nr = (d == 0) ? 2 : 1;
            size_t scOff = 0, veOff = 0;
            size_t scOffs[2]; float* vePtr[2];

            // phase 1: relation GEMMs + raw scores + segment max
            for (int j = 0; j < nr; ++j) {
                int r = dstEdges[d][j], s = srcOf[r], Ns = Nn[s], E = Ecnt[r];
                float* keP = big0;                       // reused per edge type
                float* veP = veS + veOff; vePtr[j] = veP; veOff += (size_t)Ns * 64;
                gemm_bf16_n64<<<divUp(Ns, 128), B, 0, stream>>>(
                    kb[s], Ns, WS(12 + r), nullptr, keP, nullptr, 0);
                gemm_bf16_n64<<<divUp(Ns, 128), B, 0, stream>>>(
                    vb[s], Ns, WS(16 + r), nullptr, veP, nullptr, 0);
                k_edge_score<<<divUp(E * 2, 256), B, 0, stream>>>(
                    ei[r], E, qd[d], keP, L(base + prelO[r]), segmax, scB + scOff);
                scOffs[j] = scOff; scOff += (size_t)E * 2;
            }
            // phase 2: exp + denominator
            for (int j = 0; j < nr; ++j) {
                int r = dstEdges[d][j], E = Ecnt[r];
                k_edge_exp<<<divUp(E * 2, 256), B, 0, stream>>>(
                    ei[r], E, segmax, den, scB + scOffs[j]);
            }
            // phase 3: weighted aggregation
            for (int j = 0; j < nr; ++j) {
                int r = dstEdges[d][j], E = Ecnt[r];
                k_edge_agg<<<divUp(E * 2, 256), B, 0, stream>>>(
                    ei[r], E, den, scB + scOffs[j], vePtr[j], agg);
            }
            // gelu -> A-projection (WMMA) -> skip-gate + residual + LN + relu
            k_gelu<<<divUp(Nd * 64, 256), B, 0, stream>>>(agg, geluB, Nd * 64);
            gemm_bf16_n64<<<divUp(Nd, 128), B, 0, stream>>>(
                geluB, Nd, WS(9 + d), L(base + aB[d]), big0, nullptr, 0);
            k_combine_ln<<<divUp(Nd * 32, 256), B, 0, stream>>>(
                hf[d], hb[d], big0, L(base + skipO[d]), L(lnG[d]), L(lnBt[d]), Nd);
        }
    }

    // ---- heads ----
    k_cvt_swz<<<8, B, 0, stream>>>(L(head1W), WS(0));
    gemm_bf16_n64<<<divUp(NF, 128), B, 0, stream>>>(
        hb[0], NF, WS(0), L(head1b), big0, nullptr, 1 /*relu*/);
    k_head2<<<divUp(NF, 256), B, 0, stream>>>(
        big0, L(head2W), L(head2b), (float*)d_out, NF);
}
